// TimeLSTM_78761110274232
// MI455X (gfx1250) — matrix operand order repaired and checked by reading, not compiled, not run on hardware
//
#include <hip/hip_runtime.h>
#include <hip/hip_bf16.h>
#include <math.h>

// ---------------------------------------------------------------------------
// Time-LSTM for MI455X (gfx1250, wave32, WMMA 16x16x32 bf16)
// B=64, S=1024, D=128, H=256, 4H=1024
// ---------------------------------------------------------------------------

typedef __attribute__((ext_vector_type(16))) __bf16 v16bf;
typedef __attribute__((ext_vector_type(8)))  float  v8f;

#define B_   64
#define S_   1024
#define D_   128
#define H_   256
#define G4H  1024   // 4*H

__device__ __forceinline__ unsigned short f2bf(float f) {
  union { float f; unsigned u; } v; v.f = f;
  unsigned r = v.u + 0x7FFFu + ((v.u >> 16) & 1u);   // round-to-nearest-even
  return (unsigned short)(r >> 16);
}

#if __has_builtin(__builtin_amdgcn_tanhf)
__device__ __forceinline__ float fast_tanh(float x) { return __builtin_amdgcn_tanhf(x); }
#else
__device__ __forceinline__ float fast_tanh(float x) { return tanhf(x); }
#endif
__device__ __forceinline__ float fast_sigmoid(float x) {
  return 1.0f / (1.0f + __expf(-x));
}

// ---------------------------------------------------------------------------
// Fragment layouts (wave32):
//   A (16x32 MxK bf16): lane l: m=l&15, half=l>>4; elem i -> K = kt*32 +
//       ((i&8)<<1) + half*8 + (i&7)
//   B (32x16 KxN bf16): lane l: n=l&15, half=l>>4; elem i -> K = kt*32 +
//       ((i&8)<<1) + half*8 + (i&7)
//   C/D (16x16 f32):    lane l: n=l&15; elem r -> m = r + (l>>4)*8
// B operands are pre-swizzled so each lane reads 16 contiguous bf16 (32B).
// ---------------------------------------------------------------------------

// W_all [4H=1024 n][H=256 k] -> fragments [nt(64)][kt(8)][lane(32)][i(16)]
__global__ void prep_wall_frags(const float* __restrict__ W,
                                unsigned short* __restrict__ frag) {
  int gid = blockIdx.x * blockDim.x + threadIdx.x;
  if (gid >= G4H * H_) return;
  int f = gid >> 9;
  int l = (gid >> 4) & 31;
  int i = gid & 15;
  int nt = f >> 3, kt = f & 7;
  int n = nt * 16 + (l & 15);
  int k = kt * 32 + ((i & 8) << 1) + ((l >> 4) << 3) + (i & 7);
  frag[gid] = f2bf(W[n * H_ + k]);
}

// W_d [256 n][256 k] -> fragments [nt(16)][kt(8)][lane][i]
__global__ void prep_wd_frags(const float* __restrict__ W,
                              unsigned short* __restrict__ frag) {
  int gid = blockIdx.x * blockDim.x + threadIdx.x;
  if (gid >= H_ * H_) return;
  int f = gid >> 9;
  int l = (gid >> 4) & 31;
  int i = gid & 15;
  int nt = f >> 3, kt = f & 7;
  int n = nt * 16 + (l & 15);
  int k = kt * 32 + ((i & 8) << 1) + ((l >> 4) << 3) + (i & 7);
  frag[gid] = f2bf(W[n * H_ + k]);
}

// U_all [1024 n][128 k] -> fragments [nt(64)][kt(4)][lane][i]
__global__ void prep_u_frags(const float* __restrict__ U,
                             unsigned short* __restrict__ frag) {
  int gid = blockIdx.x * blockDim.x + threadIdx.x;
  if (gid >= G4H * D_) return;
  int f = gid >> 9;
  int l = (gid >> 4) & 31;
  int i = gid & 15;
  int nt = f >> 2, kt = f & 3;
  int n = nt * 16 + (l & 15);
  int k = kt * 32 + ((i & 8) << 1) + ((l >> 4) << 3) + (i & 7);
  frag[gid] = f2bf(U[n * D_ + k]);
}

// decay stored TIME-MAJOR: decay_t[s*B + b]
__global__ void compute_decay(const float* __restrict__ ts,
                              float* __restrict__ decay_t) {
  int gid = blockIdx.x * blockDim.x + threadIdx.x;
  if (gid >= B_ * S_) return;
  int b = gid >> 10;            // ts is [B][S]
  int s = gid & (S_ - 1);
  float d = 1.0f;
  if (s != 0) {
    float dt = fmaxf(ts[gid] - ts[gid - 1], 1e-6f);
    d = 1.0f / logf(2.718281828459045f + dt);
  }
  decay_t[s * B_ + b] = d;
}

// ---------------------------------------------------------------------------
// u = inputs @ U_allT + U_all_b + W_all_b (W_all bias folded in).
// Stored TIME-MAJOR: u[s][b][4H] so the scan streams one contiguous 64 KB
// chunk per block per step.
// grid (4096 M-tiles, 8 N-groups), block 128 = 4 waves, 2 N-tiles per wave.
// ---------------------------------------------------------------------------
__global__ __launch_bounds__(128) void input_proj(
    const float* __restrict__ x, const unsigned short* __restrict__ uf,
    const float* __restrict__ ub, const float* __restrict__ wb,
    float* __restrict__ u) {
  __shared__ unsigned short afrag[4 * 512];
  const int mt  = blockIdx.x;
  const int ng  = blockIdx.y;
  const int tid = threadIdx.x;

  // stage + swizzle the 16x128 input tile as bf16 A-fragments
  for (int idx = tid; idx < 2048; idx += 128) {
    int kt = idx >> 9, rem = idx & 511;
    int l = rem >> 4, i = rem & 15;
    int m = l & 15;
    int k = kt * 32 + ((i & 8) << 1) + ((l >> 4) << 3) + (i & 7);
    afrag[idx] = f2bf(x[(mt * 16 + m) * D_ + k]);
  }
  __syncthreads();

  const int w = tid >> 5, lane = tid & 31;
  const int half = lane >> 4;
#pragma unroll
  for (int q = 0; q < 2; ++q) {
    const int nt = ng * 8 + w * 2 + q;
    v8f acc = {};
#pragma unroll
    for (int kt = 0; kt < 4; ++kt) {
      v16bf a = *(const v16bf*)&afrag[kt * 512 + lane * 16];
      v16bf b = *(const v16bf*)&uf[(nt * 4 + kt) * 512 + lane * 16];
      acc = __builtin_amdgcn_wmma_f32_16x16x32_bf16(false, a, false, b,
                                                    (short)0, acc, false, false);
    }
    const int n = nt * 16 + (lane & 15);
    const float bias = ub[n] + wb[n];
#pragma unroll
    for (int r = 0; r < 8; ++r) {
      int m = r + half * 8;
      int grow = mt * 16 + m;               // = b*S + s
      int b = grow >> 10, s = grow & (S_ - 1);
      u[((size_t)s * B_ + b) * G4H + n] = acc[r] + bias;
    }
  }
}

// ---------------------------------------------------------------------------
// Sequential scan. grid = 4 blocks (16 batch rows each), block = 512 = 16
// waves. Wave w owns hidden columns [16w, 16w+16): computes the W_d tile and
// the four gate tiles for those columns. h/c state lives in REGISTERS per
// lane; updates write bf16 directly into the swizzled LDS fragment slots, so
// there is no per-step fragment rebuild. W_d fragments are hoisted into
// VGPRs; W_all streams from L2; next step's u chunk is prefetched.
// ---------------------------------------------------------------------------
__global__ __launch_bounds__(512) void scan_kernel(
    const unsigned short* __restrict__ wallf,
    const unsigned short* __restrict__ wdf,
    const float* __restrict__ wd_b,
    const float* __restrict__ u,        // [S][B][4H], biases folded in
    const float* __restrict__ decay_t,  // [S][B]
    float* __restrict__ out,            // [B,S,H]
    float* __restrict__ h_out,          // [B,H]
    float* __restrict__ c_out) {        // [B,H]
  __shared__ unsigned short hfrag[8 * 512];
  __shared__ unsigned short cfrag[8 * 512];

  const int tid  = threadIdx.x;
  const int w    = tid >> 5;
  const int lane = tid & 31;
  const int b0   = blockIdx.x * 16;
  const int half = lane >> 4;
  const int nloc = lane & 15;
  const int col  = w * 16 + nloc;       // hidden column owned by this lane

  // zero-init fragments (h = c = 0)
  for (int idx = tid; idx < 4096; idx += 512) { hfrag[idx] = 0; cfrag[idx] = 0; }

  // register-resident recurrent state: rows m = r + half*8, column `col`
  float creg[8], hreg[8];
#pragma unroll
  for (int r = 0; r < 8; ++r) { creg[r] = 0.f; hreg[r] = 0.f; }

  // hoist W_d fragments for N-tile w into VGPRs (reused 1024x)
  v16bf wdreg[8];
#pragma unroll
  for (int kt = 0; kt < 8; ++kt)
    wdreg[kt] = *(const v16bf*)&wdf[(w * 8 + kt) * 512 + lane * 16];

  const float wdb = wd_b[col];

  // fragment slot for element (m, k=col): lane-constant base + m*16
  const int kt_w = col >> 5;
  const int kloc = col & 31;
  const int fh   = (kloc >> 3) & 1;
  const int fi   = (((kloc >> 4) & 1) << 3) | (kloc & 7);
  const int slotbase = kt_w * 512 + fh * 256 + fi;

  __syncthreads();

  for (int s = 0; s < S_; ++s) {
    // prefetch next step's contiguous 64 KB u chunk (this block's 16 rows)
    if (s + 1 < S_) {
      const char* nxt = (const char*)&u[((size_t)(s + 1) * B_ + b0) * G4H];
      __builtin_prefetch(nxt + tid * 128, 0, 0);
    }

    // c_short pre-activation: c @ WdT (N-tile w), W_d from registers
    v8f csacc = {};
#pragma unroll
    for (int kt = 0; kt < 8; ++kt) {
      v16bf a = *(const v16bf*)&cfrag[kt * 512 + lane * 16];
      csacc = __builtin_amdgcn_wmma_f32_16x16x32_bf16(false, a, false, wdreg[kt],
                                                      (short)0, csacc, false, false);
    }

    // gate pre-activations: h @ WallT, N-tiles { g*16 + w }, W_all from L2
    v8f gacc[4];
#pragma unroll
    for (int g = 0; g < 4; ++g) {
      v8f acc = {};
      const int ntbase = (g * 16 + w) * 8;
#pragma unroll
      for (int kt = 0; kt < 8; ++kt) {
        v16bf a = *(const v16bf*)&hfrag[kt * 512 + lane * 16];
        v16bf b = *(const v16bf*)&wallf[(ntbase + kt) * 512 + lane * 16];
        acc = __builtin_amdgcn_wmma_f32_16x16x32_bf16(false, a, false, b,
                                                      (short)0, acc, false, false);
      }
      gacc[g] = acc;
    }
    __syncthreads();   // all fragment reads done before overwriting

    // elementwise Time-LSTM update; write next-step fragments in place
    const size_t ustep = (size_t)s * B_;
#pragma unroll
    for (int r = 0; r < 8; ++r) {
      const int m = r + half * 8;
      const size_t urow = (ustep + (b0 + m)) * G4H;
      const float dtv    = decay_t[ustep + (b0 + m)];
      const float cshort = fast_tanh(csacc[r] + wdb);
      const float cadj   = (creg[r] - cshort) + cshort * dtv;
      const float fg = fast_sigmoid(gacc[0][r] + u[urow + 0 * H_ + col]);
      const float ig = fast_sigmoid(gacc[1][r] + u[urow + 1 * H_ + col]);
      const float og = fast_sigmoid(gacc[2][r] + u[urow + 2 * H_ + col]);
      const float ct = fast_tanh   (gacc[3][r] + u[urow + 3 * H_ + col]);
      const float cn = fg * cadj + ig * ct;
      const float hn = og * fast_tanh(cn);
      creg[r] = cn;
      hreg[r] = hn;
      const int slot = slotbase + m * 16;
      cfrag[slot] = f2bf(cn);
      hfrag[slot] = f2bf(hn);
      out[((size_t)(b0 + m) * S_ + s) * H_ + col] = hn;
    }
    __syncthreads();   // fragment writes visible before next step's GEMMs
  }

  // final h, c from registers
#pragma unroll
  for (int r = 0; r < 8; ++r) {
    const int m = r + half * 8;
    h_out[(b0 + m) * H_ + col] = hreg[r];
    c_out[(b0 + m) * H_ + col] = creg[r];
  }
}

// ---------------------------------------------------------------------------
extern "C" void kernel_launch(void* const* d_in, const int* in_sizes, int n_in,
                              void* d_out, int out_size, void* d_ws, size_t ws_size,
                              hipStream_t stream) {
  const float* inputs = (const float*)d_in[0];
  const float* ts     = (const float*)d_in[1];
  const float* Wall_w = (const float*)d_in[2];
  const float* Wall_b = (const float*)d_in[3];
  const float* U_w    = (const float*)d_in[4];
  const float* U_b    = (const float*)d_in[5];
  const float* Wd_w   = (const float*)d_in[6];
  const float* Wd_b   = (const float*)d_in[7];

  char* ws = (char*)d_ws;
  unsigned short* wallf = (unsigned short*)(ws);               // 512 KB
  unsigned short* wdf   = (unsigned short*)(ws + 524288);      // 128 KB
  unsigned short* uf    = (unsigned short*)(ws + 655360);      // 256 KB
  float*          decay = (float*)(ws + 917504);               // 256 KB
  float*          u     = (float*)(ws + 1179648);              // 256 MB

  float* out   = (float*)d_out;
  float* h_out = out + (size_t)B_ * S_ * H_;
  float* c_out = h_out + (size_t)B_ * H_;

  hipLaunchKernelGGL(prep_wall_frags, dim3(1024), dim3(256), 0, stream, Wall_w, wallf);
  hipLaunchKernelGGL(prep_wd_frags,   dim3(256),  dim3(256), 0, stream, Wd_w, wdf);
  hipLaunchKernelGGL(prep_u_frags,    dim3(512),  dim3(256), 0, stream, U_w, uf);
  hipLaunchKernelGGL(compute_decay,   dim3(256),  dim3(256), 0, stream, ts, decay);
  hipLaunchKernelGGL(input_proj, dim3(4096, 8), dim3(128), 0, stream,
                     inputs, uf, U_b, Wall_b, u);
  hipLaunchKernelGGL(scan_kernel, dim3(4), dim3(512), 0, stream,
                     wallf, wdf, Wd_b, u, decay, out, h_out, c_out);
}
